// Hierarchy_Embedding_11793980195160
// MI455X (gfx1250) — matrix operand order repaired and checked by reading, not compiled, not run on hardware
//
#include <hip/hip_runtime.h>

// ---------------------------------------------------------------------------
// Hierarchy embedding for MI455X (gfx1250). All heavy math routed through
// v_wmma_f32_16x16x32_bf16 (wave32). GEMM uses 64x64 workgroup tiles with
// double-buffered LDS staging via async global->LDS loads (ASYNCcnt).
// One-hot hierarchy matrices are collapsed to parent-index arrays and applied
// as softmax predicates.
// ---------------------------------------------------------------------------

typedef __bf16 bf16;
typedef __attribute__((ext_vector_type(16))) __bf16 bf16x16;
typedef __attribute__((ext_vector_type(8)))  __bf16 bf16x8;
typedef __attribute__((ext_vector_type(8)))  float  f32x8;

#define CN1 64
#define CN2 512
#define CN3 2048
#define CD  512
#define CB  16
#define CN  512
#define CNH 8
#define CHD 64
#define APAD 40   // LDS row stride (elements) for 32-elem K-slice, padded

#if defined(__has_builtin)
#if __has_builtin(__builtin_amdgcn_global_load_async_to_lds_b128) && \
    __has_builtin(__builtin_amdgcn_s_wait_asynccnt)
#define USE_ASYNC_LDS 1
#endif
#endif

#ifdef USE_ASYNC_LDS
typedef int v4i __attribute__((vector_size(16)));
typedef __attribute__((address_space(1))) v4i* as1_v4i;
typedef __attribute__((address_space(3))) v4i* as3_v4i;
#endif

// ---------------- fragment loaders (per CDNA5 ISA layouts) -------
// A (16x32 bf16, MxK): lanes 0-15 row M=lane, K=0..7 & 16..23; lanes 16-31
// same rows, K=8..15 & 24..31.
__device__ __forceinline__ bf16x16 load_a_frag(const bf16* p) {
  bf16x8 lo = *(const bf16x8*)(p);
  bf16x8 hi = *(const bf16x8*)(p + 16);
  bf16x16 r;
#pragma unroll
  for (int i = 0; i < 8; ++i) { r[i] = lo[i]; r[i + 8] = hi[i]; }
  return r;
}
// B (32x16 bf16, KxN) from row-major N x K: lane&15 = N column, lanes 0-15
// hold K=0..7 & 16..23, lanes 16-31 hold K=8..15 & 24..31.
__device__ __forceinline__ bf16x16 load_b_frag(const bf16* p) {
  bf16x8 lo = *(const bf16x8*)(p);
  bf16x8 hi = *(const bf16x8*)(p + 16);
  bf16x16 r;
#pragma unroll
  for (int i = 0; i < 8; ++i) { r[i] = lo[i]; r[i + 8] = hi[i]; }
  return r;
}

// ---------------- WMMA GEMM: C(MxN,f32) = A(MxK)*B(NxK)^T (+bias) -----------
// grid = (M/64, N/64), block = 128 (4 waves). Each wave computes a 16x64
// slice (4 accumulators), sharing LDS-staged A/B tiles (double buffered).
__global__ __launch_bounds__(128)
void k_gemm_bf16(const bf16* __restrict__ A, int lda,
                 const bf16* __restrict__ B, int ldb,
                 float* __restrict__ C, int ldc, int K,
                 const float* __restrict__ bias, float alpha) {
  __shared__ bf16 shA[2][64 * APAD];
  __shared__ bf16 shB[2][64 * APAD];
  const int tid  = threadIdx.x;
  const int wv   = tid >> 5;
  const int lane = tid & 31;
  const int m0 = blockIdx.x * 64;
  const int n0 = blockIdx.y * 64;

  // staging: 128 threads cover 64 rows x 32 K-elems (each thread: 16 elems)
  const int srow = tid >> 1;
  const int scol = (tid & 1) * 16;
  const bf16* gA = A + (size_t)(m0 + srow) * lda + scol;
  const bf16* gB = B + (size_t)(n0 + srow) * ldb + scol;
  const int soff = srow * APAD + scol;

  auto stage = [&](int buf, int k) {
    const bf16* ga = gA + k;
    const bf16* gb = gB + k;
    bf16* sa = &shA[buf][soff];
    bf16* sb = &shB[buf][soff];
#ifdef USE_ASYNC_LDS
    __builtin_amdgcn_global_load_async_to_lds_b128((as1_v4i)(void*)ga,
                                                   (as3_v4i)(void*)sa, 0, 0);
    __builtin_amdgcn_global_load_async_to_lds_b128((as1_v4i)(void*)(ga + 8),
                                                   (as3_v4i)(void*)(sa + 8), 0, 0);
    __builtin_amdgcn_global_load_async_to_lds_b128((as1_v4i)(void*)gb,
                                                   (as3_v4i)(void*)sb, 0, 0);
    __builtin_amdgcn_global_load_async_to_lds_b128((as1_v4i)(void*)(gb + 8),
                                                   (as3_v4i)(void*)(sb + 8), 0, 0);
#else
    *(bf16x8*)sa       = *(const bf16x8*)ga;
    *(bf16x8*)(sa + 8) = *(const bf16x8*)(ga + 8);
    *(bf16x8*)sb       = *(const bf16x8*)gb;
    *(bf16x8*)(sb + 8) = *(const bf16x8*)(gb + 8);
#endif
  };

  const int fr  = lane & 15;
  const int akh = (lane >> 4) * 8;
  const int bkh = (lane >> 4) * 8;
  const f32x8 z8 = {0.f, 0.f, 0.f, 0.f, 0.f, 0.f, 0.f, 0.f};
  f32x8 acc[4] = {z8, z8, z8, z8};

  int cur = 0;
  stage(0, 0);
  for (int k = 0; k < K; k += 32) {
#ifdef USE_ASYNC_LDS
    __builtin_amdgcn_s_wait_asynccnt(0);
#endif
    __syncthreads();
    if (k + 32 < K) stage(cur ^ 1, k + 32);
    const bf16* pa = &shA[cur][(wv * 16 + fr) * APAD + akh];
    bf16x16 af = load_a_frag(pa);
#pragma unroll
    for (int j = 0; j < 4; ++j) {
      const bf16* pb = &shB[cur][(j * 16 + fr) * APAD + bkh];
      bf16x16 bfr = load_b_frag(pb);
      acc[j] = __builtin_amdgcn_wmma_f32_16x16x32_bf16(false, af, false, bfr,
                                                       (short)0, acc[j], false, false);
    }
    cur ^= 1;
  }

  // C/D layout: lanes 0-15 -> N=lane, M=vgpr; lanes 16-31 -> N=lane-16, M=vgpr+8
  const int cn = lane & 15;
  const int mb = (lane >> 4) * 8;
#pragma unroll
  for (int j = 0; j < 4; ++j) {
    const int col = n0 + j * 16 + cn;
    const float bv = bias ? bias[col] : 0.f;
#pragma unroll
    for (int rr = 0; rr < 8; ++rr) {
      C[(size_t)(m0 + wv * 16 + mb + rr) * ldc + col] = acc[j][rr] * alpha + bv;
    }
  }
}

// ---------------- helpers ---------------------------------------------------
__global__ void k_f32_to_bf16(const float* __restrict__ s, bf16* __restrict__ d,
                              size_t n) {
  size_t i = (size_t)blockIdx.x * blockDim.x + threadIdx.x;
  if (i < n) d[i] = (bf16)s[i];
}

// dst (C x R, bf16) = transpose of src (R x C, f32)
__global__ void k_trans_f32_to_bf16(const float* __restrict__ s,
                                    bf16* __restrict__ d, int R, int C) {
  size_t i = (size_t)blockIdx.x * blockDim.x + threadIdx.x;
  if (i >= (size_t)R * C) return;
  int r = (int)(i / C), c = (int)(i % C);
  d[(size_t)c * R + r] = (bf16)s[i];
}

// parent index of each column of a one-hot matrix H (rows x cols)
__global__ void k_parents(const float* __restrict__ H, int rows, int cols,
                          int* __restrict__ par) {
  int j = blockIdx.x * blockDim.x + threadIdx.x;
  if (j >= cols) return;
  int p = 0;
  for (int i = 0; i < rows; ++i)
    if (H[(size_t)i * cols + j] > 0.5f) { p = i; break; }
  par[j] = p;
}

// Row softmax with predicate masks. mode 0: none. mode 1: par[row]==par[col].
// mode 2: hierarchical keep over [up | self | children]. mode 3: byte mask.
__global__ void k_row_softmax(float* __restrict__ M, int cols, int ld, int mode,
                              const int* __restrict__ par, int upLen, int selfLen,
                              const int* __restrict__ parQ,
                              const int* __restrict__ childPar,
                              const unsigned char* __restrict__ bmask,
                              float scale) {
  const int row = blockIdx.x;
  float* rp = M + (size_t)row * ld;
  __shared__ float red[256];
  float mx = -__builtin_inff();
  for (int j = threadIdx.x; j < cols; j += blockDim.x) {
    bool ok = true;
    if (mode == 1) {
      ok = (par[row] == par[j]);
    } else if (mode == 2) {
      if (j < upLen)                ok = (parQ[row] == j);
      else if (j < upLen + selfLen) ok = ((j - upLen) == row);
      else                          ok = (childPar[j - upLen - selfLen] == row);
    } else if (mode == 3) {
      ok = (bmask[j] != 0);
    }
    float v = ok ? rp[j] * scale : -__builtin_inff();
    rp[j] = v;
    mx = fmaxf(mx, v);
  }
  red[threadIdx.x] = mx;
  __syncthreads();
  for (int s = blockDim.x >> 1; s > 0; s >>= 1) {
    if (threadIdx.x < s) red[threadIdx.x] = fmaxf(red[threadIdx.x], red[threadIdx.x + s]);
    __syncthreads();
  }
  mx = red[0];
  __syncthreads();
  float sum = 0.f;
  for (int j = threadIdx.x; j < cols; j += blockDim.x) {
    float e = __expf(rp[j] - mx);   // exp(-inf - mx) -> 0
    rp[j] = e;
    sum += e;
  }
  red[threadIdx.x] = sum;
  __syncthreads();
  for (int s = blockDim.x >> 1; s > 0; s >>= 1) {
    if (threadIdx.x < s) red[threadIdx.x] += red[threadIdx.x + s];
    __syncthreads();
  }
  const float inv = 1.f / red[0];
  for (int j = threadIdx.x; j < cols; j += blockDim.x) rp[j] *= inv;
}

// out = w[0]*a + w[1]*b + w[2]*c
__global__ void k_combine3(const float* __restrict__ a, const float* __restrict__ b,
                           const float* __restrict__ c, const float* __restrict__ w,
                           float* __restrict__ out, int n) {
  int i = blockIdx.x * blockDim.x + threadIdx.x;
  if (i < n) out[i] = w[0] * a[i] + w[1] * b[i] + w[2] * c[i];
}

// features[k,d] = fw0*r0[p1[p2[k]],d] + fw1*r1[p2[k],d] + fw2*r2[k,d]
__global__ void k_features(const float* __restrict__ rAll,
                           const int* __restrict__ p1, const int* __restrict__ p2,
                           const float* __restrict__ fw,
                           float* __restrict__ feat, bf16* __restrict__ featb) {
  int idx = blockIdx.x * blockDim.x + threadIdx.x;
  if (idx >= CN3 * CD) return;
  int k = idx >> 9, d = idx & (CD - 1);
  int pk = p2[k];
  int ppk = p1[pk];
  const float* r0 = rAll;
  const float* r1 = rAll + (size_t)CN1 * CD;
  const float* r2 = rAll + (size_t)(CN1 + CN2) * CD;
  float v = fw[0] * r0[(size_t)ppk * CD + d] + fw[1] * r1[(size_t)pk * CD + d] +
            fw[2] * r2[idx];
  feat[idx]  = v;
  featb[idx] = (bf16)v;
}

// ---------------------------------------------------------------------------
extern "C" void kernel_launch(void* const* d_in, const int* in_sizes, int n_in,
                              void* d_out, int out_size, void* d_ws, size_t ws_size,
                              hipStream_t stream) {
  (void)in_sizes; (void)n_in; (void)out_size; (void)ws_size;

  const float*         inputs = (const float*)d_in[0];
  const unsigned char* masks  = (const unsigned char*)d_in[1];
  const float*         H0     = (const float*)d_in[2];
  const float*         H1     = (const float*)d_in[3];
  const float* emb[3] = {(const float*)d_in[4], (const float*)d_in[5],
                         (const float*)d_in[6]};
  const float* fusedw = (const float*)d_in[7];
  const float *inw[3], *inb[3], *outw[3], *outb[3], *stw[3];
  for (int l = 0; l < 3; ++l) {
    inw[l]  = (const float*)d_in[8 + 5 * l];
    inb[l]  = (const float*)d_in[9 + 5 * l];
    outw[l] = (const float*)d_in[10 + 5 * l];
    outb[l] = (const float*)d_in[11 + 5 * l];
    stw[l]  = (const float*)d_in[12 + 5 * l];
  }
  float* feat_out = (float*)d_out;                         // (N3, D)
  float* attn_out = (float*)d_out + (size_t)CN3 * CD;      // (B, N3, D)

  // -------- bump allocator over d_ws --------
  char* ws = (char*)d_ws;
  size_t off = 0;
  auto alloc = [&](size_t bytes) -> void* {
    void* p = ws + off;
    off += (bytes + 255) & ~(size_t)255;
    return p;
  };
  const int MMAX = 2624;   // max kv rows (level 1)
  int*  p1    = (int*)alloc(CN2 * sizeof(int));
  int*  p2    = (int*)alloc(CN3 * sizeof(int));
  bf16* kvAll = (bf16*)alloc((size_t)MMAX * CD * 2);           // [e0;e1;e2] bf16
  bf16* inwb  = (bf16*)alloc((size_t)3 * CD * CD * 2);
  bf16* outwb = (bf16*)alloc((size_t)CD * CD * 2);
  float* Qf   = (float*)alloc((size_t)CN3 * CD * 4);
  bf16*  Qb   = (bf16*)alloc((size_t)CN3 * CD * 2);
  float* Kf   = (float*)alloc((size_t)MMAX * CD * 4);
  bf16*  Kb   = (bf16*)alloc((size_t)MMAX * CD * 2);
  float* Vf   = (float*)alloc((size_t)MMAX * CD * 4);
  bf16*  Vt   = (bf16*)alloc((size_t)CD * MMAX * 2);           // V^T (D x m)
  float* adpF = (float*)alloc((size_t)CN3 * CN3 * 4);
  bf16*  adpB = (bf16*)alloc((size_t)CN3 * CN3 * 2);
  bf16*  ht   = (bf16*)alloc((size_t)CD * CN3 * 2);            // h^T (D x n)
  float* aggF = (float*)alloc((size_t)CN3 * CD * 4);
  float* scF  = (float*)alloc((size_t)CN3 * MMAX * 4);         // one head
  bf16*  scB  = (bf16*)alloc((size_t)CN3 * MMAX * 2);
  float* OhF  = (float*)alloc((size_t)CN3 * CD * 4);
  bf16*  OhB  = (bf16*)alloc((size_t)CN3 * CD * 2);
  float* oF   = (float*)alloc((size_t)CN3 * CD * 4);
  float* rAll = (float*)alloc((size_t)MMAX * CD * 4);          // r0|r1|r2
  bf16*  featb = (bf16*)alloc((size_t)CN3 * CD * 2);
  bf16*  Xb   = (bf16*)alloc((size_t)CB * CN * CD * 2);        // inputs bf16
  float* attF = (float*)alloc((size_t)CN3 * CN * 4);           // one batch
  bf16*  attB = (bf16*)alloc((size_t)CN3 * CN * 2);
  bf16*  Xt   = (bf16*)alloc((size_t)CD * CN * 2);             // X_b^T

  auto cvt = [&](const float* s, bf16* d, size_t n) {
    k_f32_to_bf16<<<dim3((unsigned)((n + 255) / 256)), dim3(256), 0, stream>>>(s, d, n);
  };
  auto trans = [&](const float* s, bf16* d, int R, int C) {
    size_t n = (size_t)R * C;
    k_trans_f32_to_bf16<<<dim3((unsigned)((n + 255) / 256)), dim3(256), 0, stream>>>(s, d, R, C);
  };
  auto gemm = [&](const bf16* A, int lda, const bf16* B, int ldb, float* C,
                  int ldc, int M, int N, int K, const float* bias) {
    k_gemm_bf16<<<dim3(M / 64, N / 64), dim3(128), 0, stream>>>(A, lda, B, ldb, C,
                                                                ldc, K, bias, 1.0f);
  };
  auto softmax = [&](float* M, int rows, int cols, int ld, int mode,
                     const int* par, int upLen, int selfLen, const int* parQ,
                     const int* chp, const unsigned char* bm, float scale) {
    k_row_softmax<<<dim3(rows), dim3(256), 0, stream>>>(M, cols, ld, mode, par,
                                                        upLen, selfLen, parQ, chp,
                                                        bm, scale);
  };

  // -------- preprocessing --------
  k_parents<<<dim3(2), dim3(256), 0, stream>>>(H0, CN1, CN2, p1);
  k_parents<<<dim3(8), dim3(256), 0, stream>>>(H1, CN2, CN3, p2);
  cvt(emb[0], kvAll,                            (size_t)CN1 * CD);
  cvt(emb[1], kvAll + (size_t)CN1 * CD,         (size_t)CN2 * CD);
  cvt(emb[2], kvAll + (size_t)(CN1 + CN2) * CD, (size_t)CN3 * CD);
  cvt(inputs, Xb, (size_t)CB * CN * CD);

  // -------- per-level refinement --------
  const int nlv[3]   = {CN1, CN2, CN3};
  const int hrow[3]  = {0, CN1, CN1 + CN2};
  const int kvrow[3] = {0, 0, CN1};
  const int mlv[3]   = {CN1 + CN2, CN1 + CN2 + CN3, CN2 + CN3};
  const int upL[3]   = {0, CN1, CN2};
  const int* parQ[3] = {nullptr, p1, p2};
  const int* chP[3]  = {p1, p2, nullptr};
  const int adpMode[3] = {0, 1, 1};
  const int* adpPar[3] = {nullptr, p1, p2};

  for (int l = 0; l < 3; ++l) {
    const int n = nlv[l], m = mlv[l];
    const bf16* hb  = kvAll + (size_t)hrow[l] * CD;
    const bf16* kvb = kvAll + (size_t)kvrow[l] * CD;
    cvt(inw[l], inwb, (size_t)3 * CD * CD);
    cvt(outw[l], outwb, (size_t)CD * CD);

    // agg = softmax(mask(h h^T)) @ h
    gemm(hb, CD, hb, CD, adpF, n, n, n, CD, nullptr);
    softmax(adpF, n, n, n, adpMode[l], adpPar[l], 0, 0, nullptr, nullptr, nullptr, 1.f);
    cvt(adpF, adpB, (size_t)n * n);
    trans(emb[l], ht, n, CD);
    gemm(adpB, n, ht, n, aggF, CD, n, CD, n, nullptr);

    // Q, K, V projections
    gemm(hb, CD, inwb, CD, Qf, CD, n, CD, CD, inb[l]);
    cvt(Qf, Qb, (size_t)n * CD);
    gemm(kvb, CD, inwb + (size_t)CD * CD, CD, Kf, CD, m, CD, CD, inb[l] + CD);
    cvt(Kf, Kb, (size_t)m * CD);
    gemm(kvb, CD, inwb + (size_t)2 * CD * CD, CD, Vf, CD, m, CD, CD, inb[l] + 2 * CD);
    trans(Vf, Vt, m, CD);

    // per-head attention (reuse one score buffer)
    for (int h = 0; h < CNH; ++h) {
      gemm(Qb + h * CHD, CD, Kb + h * CHD, CD, scF, m, n, m, CHD, nullptr);
      softmax(scF, n, m, m, 2, nullptr, upL[l], n, parQ[l], chP[l], nullptr, 0.125f);
      cvt(scF, scB, (size_t)n * m);
      gemm(scB, m, Vt + (size_t)h * CHD * m, m, OhF + h * CHD, CD, n, CHD, m, nullptr);
    }
    cvt(OhF, OhB, (size_t)n * CD);
    gemm(OhB, CD, outwb, CD, oF, CD, n, CD, CD, outb[l]);

    // r_l = w0*agg + w1*h + w2*o
    k_combine3<<<dim3((n * CD + 255) / 256), dim3(256), 0, stream>>>(
        aggF, emb[l], oF, stw[l], rAll + (size_t)hrow[l] * CD, n * CD);
  }

  // -------- features = fw0*r0[p1[p2[k]]] + fw1*r1[p2[k]] + fw2*r2[k] --------
  k_features<<<dim3((CN3 * CD + 255) / 256), dim3(256), 0, stream>>>(
      rAll, p1, p2, fusedw, feat_out, featb);

  // -------- final batched attention over inputs --------
  for (int b = 0; b < CB; ++b) {
    const bf16* Xbb = Xb + (size_t)b * CN * CD;
    // att = features @ X_b^T  (N3 x N)
    gemm(featb, CD, Xbb, CD, attF, CN, CN3, CN, CD, nullptr);
    softmax(attF, CN3, CN, CN, 3, nullptr, 0, 0, nullptr, nullptr,
            masks + (size_t)b * CN, 1.f);
    cvt(attF, attB, (size_t)CN3 * CN);
    trans(inputs + (size_t)b * CN * CD, Xt, CN, CD);
    // attn_out_b = att @ X_b  (N3 x D)
    gemm(attB, CN, Xt, CN, attn_out + (size_t)b * CN3 * CD, CD, CN3, CD, CN, nullptr);
  }
}
